// SA_Module_91070486544853
// MI455X (gfx1250) — matrix operand order repaired and checked by reading, not compiled, not run on hardware
//
#include <hip/hip_runtime.h>
#include <stdint.h>

typedef __attribute__((ext_vector_type(16))) __bf16 v16bf;
typedef __attribute__((ext_vector_type(2)))  __bf16 v2bf;
typedef __attribute__((ext_vector_type(8)))  float  v8f;

#define B_DIM   2
#define C_DIM   320
#define N_DIM   4096
#define NTILE   32              // rows of attention per block (2 WMMA n-frags)
#define KCHUNK  32
#define NWAVES  10
#define BLOCK   (NWAVES * 32)   // 320 threads = 10 wave32
#define NCTILE  (C_DIM / 16)    // 20 c-tiles of 16 channels

union Frag {
    uint32_t u[8];
    v16bf    v;
};

__device__ __forceinline__ uint32_t pack_bf16(float a, float b) {
    union { v2bf v; uint32_t u; } cvt;
    cvt.v[0] = (__bf16)a;       // native v_cvt path, not bit-twiddled VALU
    cvt.v[1] = (__bf16)b;
    return cvt.u;
}

__device__ __forceinline__ float lrelu(float x) {
    return fmaxf(x, 0.01f * x);
}

// ---------------------------------------------------------------------------
// Kernel 1: s1[b,n] = sum_c p[b,c,n]*W[c]; s2[b,n] = sum_c p[b,c,n]*W[C+c]
// ---------------------------------------------------------------------------
__global__ void sa_stats(const float* __restrict__ x, const float* __restrict__ W,
                         float* __restrict__ s1, float* __restrict__ s2) {
    int idx = blockIdx.x * blockDim.x + threadIdx.x;     // 0 .. B*N-1
    int b = idx >> 12;
    int n = idx & (N_DIM - 1);
    const float* p = x + (size_t)b * C_DIM * N_DIM + n;
    float a1 = 0.f, a2 = 0.f;
    for (int c = 0; c < C_DIM; ++c) {
        float v = p[(size_t)c * N_DIM];
        a1 = fmaf(v, W[c], a1);
        a2 = fmaf(v, W[C_DIM + c], a2);
    }
    s1[idx] = a1;
    s2[idx] = a2;
}

// ---------------------------------------------------------------------------
// Kernel 2: fused softmax(lrelu(rank-1 energy)) + 320x4096x4096 bf16 WMMA GEMM
// One block = one batch, one 32-row n-tile (2 n-frags), all 320 channels.
// Each wave: 2 c-tiles x 2 n-frags = 4 WMMAs per k-chunk.
// ---------------------------------------------------------------------------
__global__ void __launch_bounds__(BLOCK)
sa_attn(const float* __restrict__ x, const float* __restrict__ s1g,
        const float* __restrict__ s2g, float* __restrict__ out) {
    __shared__ float sm_red[16 * BLOCK];                          // 20 KB reductions
    __shared__ __align__(16) uint32_t sm_a[NCTILE * 32 * 8];      // 20 KB: A frags (bf16)
    __shared__ __align__(16) uint32_t sm_b[2 * 32 * 8];           // 2 KB : B frags (bf16)
    __shared__ float sm_m[NTILE], sm_invz[NTILE], sm_s2[NTILE];

    const int t    = threadIdx.x;
    const int lane = t & 31;
    const int wave = t >> 5;
    const int b    = blockIdx.x >> 7;            // 128 n-tiles per batch
    const int n0   = (blockIdx.x & 127) * NTILE;

    const float* s1b = s1g + (size_t)b * N_DIM;

    // ---------------- softmax stats, processed in two 16-row halves --------
    for (int h = 0; h < 2; ++h) {
        const int r0 = h * 16;
        float s2r[16];
#pragma unroll
        for (int r = 0; r < 16; ++r) s2r[r] = s2g[(size_t)b * N_DIM + n0 + r0 + r];
        if (t < 16) sm_s2[r0 + t] = s2r[t];

        float mx[16];
#pragma unroll
        for (int r = 0; r < 16; ++r) mx[r] = -3.4e38f;
        for (int k = t; k < N_DIM; k += BLOCK) {
            float e1 = s1b[k];
#pragma unroll
            for (int r = 0; r < 16; ++r)
                mx[r] = fmaxf(mx[r], lrelu(e1 + s2r[r]));
        }
#pragma unroll
        for (int r = 0; r < 16; ++r) sm_red[r * BLOCK + t] = mx[r];
        __syncthreads();
        if (t < 16) {
            float m = -3.4e38f;
            for (int i = 0; i < BLOCK; ++i) m = fmaxf(m, sm_red[t * BLOCK + i]);
            sm_m[r0 + t] = m;
        }
        __syncthreads();
        float fm[16];
#pragma unroll
        for (int r = 0; r < 16; ++r) fm[r] = sm_m[r0 + r];

        float sum[16];
#pragma unroll
        for (int r = 0; r < 16; ++r) sum[r] = 0.f;
        for (int k = t; k < N_DIM; k += BLOCK) {
            float e1 = s1b[k];
#pragma unroll
            for (int r = 0; r < 16; ++r)
                sum[r] += __expf(lrelu(e1 + s2r[r]) - fm[r]);
        }
#pragma unroll
        for (int r = 0; r < 16; ++r) sm_red[r * BLOCK + t] = sum[r];
        __syncthreads();
        if (t < 16) {
            float z = 0.f;
            for (int i = 0; i < BLOCK; ++i) z += sm_red[t * BLOCK + i];
            sm_invz[r0 + t] = 1.0f / z;
        }
        __syncthreads();
    }

    // ---------------- main k-loop: stage to LDS, WMMA accumulate -----------
    const int ct0 = wave * 2;                    // each wave owns 2 c-tiles
    v8f acc00 = {}, acc01 = {}, acc10 = {}, acc11 = {};

    const float* psrc = x + ((size_t)(b * C_DIM + t)) * N_DIM;   // t == channel

    for (int k0 = 0; k0 < N_DIM; k0 += KCHUNK) {
        __syncthreads();

        // -- stage p[c, k0..k0+31] -> bf16 in WMMA A-fragment layout --------
        {
            const int c  = t;                    // BLOCK == C_DIM
            const int ct = c >> 4, m = c & 15;
            const float4* f4 = reinterpret_cast<const float4*>(psrc + k0);
            float v[KCHUNK];
#pragma unroll
            for (int i = 0; i < 8; ++i) {
                float4 f = f4[i];
                v[4 * i + 0] = f.x; v[4 * i + 1] = f.y;
                v[4 * i + 2] = f.z; v[4 * i + 3] = f.w;
            }
            if (k0 + KCHUNK < N_DIM)
                __builtin_prefetch(psrc + k0 + KCHUNK, 0, 3);    // near-scope prefetch

            uint32_t* rec0 = &sm_a[(ct * 32 + m) * 8];           // lane m  : K 0-7,16-23
            uint32_t* rec1 = &sm_a[(ct * 32 + m + 16) * 8];      // lane m+16: K 8-15,24-31
#pragma unroll
            for (int j = 0; j < 4; ++j) {
                rec0[j]     = pack_bf16(v[2 * j],      v[2 * j + 1]);
                rec0[4 + j] = pack_bf16(v[16 + 2 * j], v[17 + 2 * j]);
                rec1[j]     = pack_bf16(v[8 + 2 * j],  v[9 + 2 * j]);
                rec1[4 + j] = pack_bf16(v[24 + 2 * j], v[25 + 2 * j]);
            }
        }

        // -- stage unnormalized attention weights -> 2 B-fragments ----------
        if (t < 256) {
#pragma unroll
            for (int g = 0; g < 2; ++g) {
                const int pi = t + g * 256;          // 512 bf16 pairs total
                const int l  = (pi >> 3) & 31;
                const int pr = pi & 7;
                const int n  = g * 16 + (l & 15);
                const int kb = k0 + ((l >> 4) * 8) + (pr & 3) * 2 + (pr >> 2) * 16;
                const float m_n = sm_m[n];
                const float s2n = sm_s2[n];
                float wa = __expf(lrelu(s1b[kb]     + s2n) - m_n);
                float wb = __expf(lrelu(s1b[kb + 1] + s2n) - m_n);
                sm_b[g * 256 + l * 8 + pr] = pack_bf16(wa, wb);
            }
        }
        __syncthreads();

        // -- WMMA: D = A(p, 16x32 bf16) x B(attn^T, 32x16 bf16) + C ---------
        Frag fa0, fa1, fb0, fb1;
        const uint32_t* pa0 = &sm_a[(ct0 * 32 + lane) * 8];
        const uint32_t* pa1 = &sm_a[((ct0 + 1) * 32 + lane) * 8];
        const uint32_t* pb0 = &sm_b[lane * 8];
        const uint32_t* pb1 = &sm_b[256 + lane * 8];
#pragma unroll
        for (int i = 0; i < 8; ++i) {
            fa0.u[i] = pa0[i]; fa1.u[i] = pa1[i];
            fb0.u[i] = pb0[i]; fb1.u[i] = pb1[i];
        }

        acc00 = __builtin_amdgcn_wmma_f32_16x16x32_bf16(
                    false, fa0.v, false, fb0.v, (short)0, acc00, false, false);
        acc01 = __builtin_amdgcn_wmma_f32_16x16x32_bf16(
                    false, fa0.v, false, fb1.v, (short)0, acc01, false, false);
        acc10 = __builtin_amdgcn_wmma_f32_16x16x32_bf16(
                    false, fa1.v, false, fb0.v, (short)0, acc10, false, false);
        acc11 = __builtin_amdgcn_wmma_f32_16x16x32_bf16(
                    false, fa1.v, false, fb1.v, (short)0, acc11, false, false);
    }

    // ---------------- epilogue: normalize (1/Z) + residual add -------------
    const int   nn  = n0 + (lane & 15);
    const float iz0 = sm_invz[lane & 15];
    const float iz1 = sm_invz[16 + (lane & 15)];
    const int   chi = (lane >> 4) * 8;           // D layout: lanes>=16 hold M=j+8
#pragma unroll
    for (int j = 0; j < 8; ++j) {
        int c = ct0 * 16 + chi + j;
        size_t off = ((size_t)(b * C_DIM + c)) * N_DIM + nn;
        out[off]      = acc00[j] * iz0 + x[off];
        out[off + 16] = acc01[j] * iz1 + x[off + 16];
    }
#pragma unroll
    for (int j = 0; j < 8; ++j) {
        int c = (ct0 + 1) * 16 + chi + j;
        size_t off = ((size_t)(b * C_DIM + c)) * N_DIM + nn;
        out[off]      = acc10[j] * iz0 + x[off];
        out[off + 16] = acc11[j] * iz1 + x[off + 16];
    }
}

// ---------------------------------------------------------------------------
extern "C" void kernel_launch(void* const* d_in, const int* in_sizes, int n_in,
                              void* d_out, int out_size, void* d_ws, size_t ws_size,
                              hipStream_t stream) {
    const float* x = (const float*)d_in[0];   // (2,320,4,32,32) fp32
    const float* W = (const float*)d_in[1];   // (640,1) fp32
    float* out = (float*)d_out;

    float* s1 = (float*)d_ws;                 // B*N floats
    float* s2 = s1 + (size_t)B_DIM * N_DIM;   // B*N floats  (64 KB total)

    sa_stats<<<dim3((B_DIM * N_DIM) / 256), dim3(256), 0, stream>>>(x, W, s1, s2);
    sa_attn<<<dim3(B_DIM * (N_DIM / NTILE)), dim3(BLOCK), 0, stream>>>(x, s1, s2, out);
}
